// Model_33251636805848
// MI455X (gfx1250) — matrix operand order repaired and checked by reading, not compiled, not run on hardware
//
#include <hip/hip_runtime.h>
#include <hip/hip_bf16.h>
#include <math.h>

// Problem constants (from reference)
#define BS   64
#define LSEQ 100
#define D    512
#define TD   1536            // 3*D
#define NCAND (BS * (LSEQ + 1))   // 6464
#define NROWS (BS * LSEQ)         // 6400
#define MASK_VAL (-10000.0f)
#define RROWS 16             // batch rows owned by one recurrence block

typedef __bf16 bf16_t;
typedef __attribute__((ext_vector_type(16))) __bf16        v16bf;
typedef __attribute__((ext_vector_type(8)))  float         v8f;
typedef __attribute__((ext_vector_type(8)))  unsigned int  v8u;

// ---------------------------------------------------------------------------
// WMMA fragment loaders (wave32; this code's contiguous k order, k = 16h + s, same for A and B; not the ISA table order)
// A (16x32 bf16): lane m = lane&15; lanes 0-15 hold K 0-15, lanes 16-31
// hold K 16-31 (2 bf16 per dword).
// B (32x16 bf16): lane n = lane&15; lanes 0-15 hold K 0-15, lanes 16-31 K 16-31.
// rowp must point at the lane's row (row-major, K contiguous, dword aligned).
// Works for both global and LDS-resident rows (addrspace inferred).
// ---------------------------------------------------------------------------
__device__ __forceinline__ v16bf load_a16(const bf16_t* rowp, unsigned lane) {
    const unsigned* r = (const unsigned*)rowp;
    unsigned o = (lane & 16u) ? 8u : 0u;   // +16 bf16 = +8 dwords for upper half
    v8u u;
    u[0] = r[o + 0]; u[1] = r[o + 1]; u[2] = r[o + 2]; u[3] = r[o + 3];
    u[4] = r[o + 4]; u[5] = r[o + 5]; u[6] = r[o + 6]; u[7] = r[o + 7];
    return __builtin_bit_cast(v16bf, u);
}

__device__ __forceinline__ v16bf load_b16(const bf16_t* rowp, unsigned lane) {
    const unsigned* r = (const unsigned*)rowp;
    unsigned o = (lane & 16u) ? 8u : 0u;   // +16 bf16 = +8 dwords for upper half
    v8u u;
#pragma unroll
    for (int v = 0; v < 8; ++v) u[v] = r[o + v];
    return __builtin_bit_cast(v16bf, u);
}

// ---------------------------------------------------------------------------
// K-init: zero the loss accumulators
// ---------------------------------------------------------------------------
__global__ void k_init(float* __restrict__ acc) {
    if (threadIdx.x < 2) acc[threadIdx.x] = 0.0f;
}

// ---------------------------------------------------------------------------
// K0: gather candidate embeddings -> bf16, debias = log(pop), column validity
// ---------------------------------------------------------------------------
__global__ void k_gather(const int* __restrict__ ids, const float* __restrict__ pop,
                         const float* __restrict__ emb, const int* __restrict__ log_mask,
                         bf16_t* __restrict__ sbf, float* __restrict__ debias,
                         unsigned char* __restrict__ colvalid) {
    int k = blockIdx.x;                    // candidate 0..NCAND-1
    int id = ids[k];
    const float* src = emb + (size_t)id * D;
    bf16_t* dst = sbf + (size_t)k * D;
    for (int j = threadIdx.x; j < D; j += blockDim.x) dst[j] = (bf16_t)src[j];
    if (threadIdx.x == 0) {
        debias[k] = __logf(pop[id]);
        int b = k / (LSEQ + 1), j = k % (LSEQ + 1);
        colvalid[k] = (j == LSEQ) ? (unsigned char)1
                                  : (unsigned char)(log_mask[b * LSEQ + j] != 0);
    }
}

// ---------------------------------------------------------------------------
// K1: W_ih, W_hh (D x 3D f32) -> transposed bf16 (3D x D) so B-fragments are
// contiguous-dword loads
// ---------------------------------------------------------------------------
__global__ void k_convw(const float* __restrict__ wih, const float* __restrict__ whh,
                        bf16_t* __restrict__ wt_ih, bf16_t* __restrict__ wt_hh) {
    int idx = blockIdx.x * blockDim.x + threadIdx.x;  // n*D + k
    if (idx < D * TD) {
        int k = idx % D, n = idx / D;
        wt_ih[idx] = (bf16_t)wih[(size_t)k * TD + n];
        wt_hh[idx] = (bf16_t)whh[(size_t)k * TD + n];
    }
}

// ---------------------------------------------------------------------------
// K2: per-sample rejection table reject[b][k] = any_j(ids[b][j] == ids_flat[k])
// ---------------------------------------------------------------------------
__global__ void k_reject(const int* __restrict__ ids, unsigned char* __restrict__ reject) {
    __shared__ int sid[LSEQ + 1];
    int b = blockIdx.y;
    if (threadIdx.x < LSEQ + 1) sid[threadIdx.x] = ids[b * (LSEQ + 1) + threadIdx.x];
    __syncthreads();
    int k = blockIdx.x * blockDim.x + threadIdx.x;
    if (k < NCAND) {
        int id = ids[k];
        int r = 0;
#pragma unroll 4
        for (int j = 0; j < LSEQ + 1; ++j) r |= (sid[j] == id);
        reject[(size_t)b * NCAND + k] = (unsigned char)r;
    }
}

// ---------------------------------------------------------------------------
// K3: gates_x = X(6400x512) @ W_ih^T-layout + b_ih   (WMMA bf16, f32 accum)
// One wave = one 16x32 output tile (A fragment reused across 2 N tiles).
// Tiles = 400 x 48.
// ---------------------------------------------------------------------------
__global__ __launch_bounds__(256) void k_gates(const bf16_t* __restrict__ xbf,
                                               const bf16_t* __restrict__ wt,
                                               const float* __restrict__ bias,
                                               float* __restrict__ gx) {
    unsigned lane = threadIdx.x & 31u;
    int wave = threadIdx.x >> 5;
    int tile = blockIdx.x * 8 + wave;           // < 400*48
    int mt = tile / (TD / 32), nt2 = tile % (TD / 32);
    int m  = mt * 16 + (lane & 15);
    int sr = (m / LSEQ) * (LSEQ + 1) + (m % LSEQ);
    const bf16_t* arow  = xbf + (size_t)sr * D;
    const bf16_t* brow0 = wt + (size_t)(nt2 * 32 + (lane & 15)) * D;
    const bf16_t* brow1 = brow0 + (size_t)16 * D;
    v8f c0 = {}, c1 = {};
#pragma unroll
    for (int kk = 0; kk < D; kk += 32) {
        v16bf a  = load_a16(arow + kk, lane);
        v16bf b0 = load_b16(brow0 + kk, lane);
        v16bf b1 = load_b16(brow1 + kk, lane);
        c0 = __builtin_amdgcn_wmma_f32_16x16x32_bf16(false, a, false, b0, (short)0, c0,
                                                     false, false);
        c1 = __builtin_amdgcn_wmma_f32_16x16x32_bf16(false, a, false, b1, (short)0, c1,
                                                     false, false);
    }
    int ncol0 = nt2 * 32 + (int)(lane & 15);
    int mbase = (lane & 16u) ? 8 : 0;
    float bv0 = bias[ncol0], bv1 = bias[ncol0 + 16];
#pragma unroll
    for (int v = 0; v < 8; ++v) {
        int row = mt * 16 + mbase + v;
        gx[(size_t)row * TD + ncol0]      = c0[v] + bv0;
        gx[(size_t)row * TD + ncol0 + 16] = c1[v] + bv1;
    }
}

// ---------------------------------------------------------------------------
// K4: ENTIRE GRU recurrence in one launch. The recurrence is independent per
// batch row, so 4 persistent blocks (16 rows each) loop t=0..99 with only
// intra-block barriers:
//   phase 1: gh(16x1536) = hbf(LDS) @ W_hh + b_hh     (WMMA, A from LDS)
//   phase 2: gate math; h in registers; hbf(LDS) updated; prec row -> global
// gh (96KB) and hbf (16KB) never leave the WGP's LDS.
// ---------------------------------------------------------------------------
__global__ __launch_bounds__(256) void k_recur(const float* __restrict__ gx,
                                               const bf16_t* __restrict__ wt,
                                               const float* __restrict__ bias,
                                               bf16_t* __restrict__ prec) {
    __shared__ float  gh[RROWS * TD];      // 96 KB
    __shared__ bf16_t hbf[RROWS * D];      // 16 KB
    unsigned lane = threadIdx.x & 31u;
    int wave = threadIdx.x >> 5;
    int rowbase = blockIdx.x * RROWS;      // batch rows owned by this block

    // zero the LDS-resident bf16 hidden state
    for (int i = threadIdx.x; i < RROWS * D; i += 256) hbf[i] = (bf16_t)0.0f;

    // f32 master hidden state lives in registers: element it of this thread is
    // (row = (tid+256*it)>>9, j = (tid+256*it)&511) -- fixed across steps.
    float hreg[32];
#pragma unroll
    for (int it = 0; it < 32; ++it) hreg[it] = 0.0f;

    __syncthreads();

    const bf16_t* a_lds_row = hbf + (size_t)(lane & 15) * D;
    int mbase = (lane & 16u) ? 8 : 0;

    for (int t = 0; t < LSEQ; ++t) {
        // ---- phase 1: gh = hbf @ W_hh^T-layout + b_hh ----
        v16bf afr[16];
#pragma unroll
        for (int kk = 0; kk < 16; ++kk) afr[kk] = load_a16(a_lds_row + kk * 32, lane);

        for (int i = 0; i < 12; ++i) {
            int tt = wave + 8 * i;          // N tile 0..95
            int ncol = tt * 16 + (int)(lane & 15);
            const bf16_t* brow = wt + (size_t)ncol * D;
            v8f c = {};
#pragma unroll
            for (int kk = 0; kk < 16; ++kk) {
                v16bf b = load_b16(brow + kk * 32, lane);
                c = __builtin_amdgcn_wmma_f32_16x16x32_bf16(false, afr[kk], false, b,
                                                            (short)0, c, false, false);
            }
            float bv = bias[ncol];
#pragma unroll
            for (int v = 0; v < 8; ++v)
                gh[(mbase + v) * TD + ncol] = c[v] + bv;
        }
        __syncthreads();   // gh complete; everyone done reading hbf

        // ---- phase 2: GRU gate math ----
#pragma unroll
        for (int it = 0; it < 32; ++it) {
            int e = (int)threadIdx.x + 256 * it;
            int row = e >> 9, j = e & (D - 1);
            int b = rowbase + row;
            const float* gxr = gx + (size_t)(b * LSEQ + t) * TD;
            float xr = gxr[j], xz = gxr[j + D], xn = gxr[j + 2 * D];
            float hr = gh[row * TD + j];
            float hz = gh[row * TD + j + D];
            float hn = gh[row * TD + j + 2 * D];
            float r = 1.0f / (1.0f + __expf(-(xr + hr)));
            float z = 1.0f / (1.0f + __expf(-(xz + hz)));
            float n = tanhf(xn + r * hn);
            float hnew = (1.0f - z) * n + z * hreg[it];
            hreg[it] = hnew;
            hbf[row * D + j] = (bf16_t)hnew;
            prec[(size_t)(b * LSEQ + t) * D + j] = (bf16_t)hnew;
        }
        __syncthreads();   // hbf ready for next step; gh free to overwrite
    }
}

// ---------------------------------------------------------------------------
// K6: fused logits GEMM + masking + online log-softmax + NLL accumulation.
// One block = 16 rows; 8 waves split the 404 column tiles; A fragments cached
// in registers. Each lane keeps a PRIVATE running (max, sumexp) for its own
// column subset; lanes/waves merged once at the end with the flash combine.
// ---------------------------------------------------------------------------
__global__ __launch_bounds__(256) void k_logits(const bf16_t* __restrict__ prec,
                                                const bf16_t* __restrict__ sbf,
                                                const float* __restrict__ debias,
                                                const unsigned char* __restrict__ colvalid,
                                                const unsigned char* __restrict__ reject,
                                                const int* __restrict__ log_mask,
                                                float* __restrict__ acc) {
    __shared__ float smx[8][16], ssm[8][16], stg[8][16];
    unsigned lane = threadIdx.x & 31u;
    int wave = threadIdx.x >> 5;
    int rowblk = blockIdx.x;                 // 0..399
    int m = (int)(lane & 15);

    // Cache this wave's A fragments (16 rows x 512 K) in registers.
    const bf16_t* arow = prec + (size_t)(rowblk * 16 + m) * D;
    v16bf afr[16];
#pragma unroll
    for (int kk = 0; kk < 16; ++kk) afr[kk] = load_a16(arow + kk * 32, lane);

    int mbase = (lane & 16u) ? 8 : 0;
    int bs_[8], tgts[8];
#pragma unroll
    for (int v = 0; v < 8; ++v) {
        int row = rowblk * 16 + mbase + v;
        int b = row / LSEQ, t = row % LSEQ;
        bs_[v] = b;
        tgts[v] = b * (LSEQ + 1) + t + 1;
    }

    float run_mx[8], run_sm[8], run_tg[8];
#pragma unroll
    for (int v = 0; v < 8; ++v) { run_mx[v] = -INFINITY; run_sm[v] = 0.0f; run_tg[v] = -INFINITY; }

    for (int ct = wave; ct < NCAND / 16; ct += 8) {
        int gcol = ct * 16 + m;
        const bf16_t* brow = sbf + (size_t)gcol * D;
        // pull next tile's row toward the WGP while WMMAs run
        if (ct + 8 < NCAND / 16)
            __builtin_prefetch(sbf + (size_t)(gcol + 128) * D, 0, 1);
        v8f c = {};
#pragma unroll
        for (int kk = 0; kk < 16; ++kk) {
            v16bf b = load_b16(brow + kk * 32, lane);
            c = __builtin_amdgcn_wmma_f32_16x16x32_bf16(false, afr[kk], false, b,
                                                        (short)0, c, false, false);
        }
        float db = debias[gcol];
        int cv = colvalid[gcol];
#pragma unroll
        for (int v = 0; v < 8; ++v) {
            float x = c[v] - db;
            int rej = reject[(size_t)bs_[v] * NCAND + gcol];
            bool msk = (cv == 0) || (rej && (gcol != tgts[v]));
            x = msk ? MASK_VAL : x;
            if (gcol == tgts[v]) run_tg[v] = x;
            // per-lane online softmax (no cross-lane traffic here)
            float nmx = fmaxf(run_mx[v], x);
            run_sm[v] = run_sm[v] * __expf(run_mx[v] - nmx) + __expf(x - nmx);
            run_mx[v] = nmx;
        }
    }

    // Merge the 16 per-lane partials per row (flash combine) + target bcast.
#pragma unroll
    for (int v = 0; v < 8; ++v) {
        float mx = run_mx[v], sm = run_sm[v], tg = run_tg[v];
#pragma unroll
        for (int s = 1; s <= 8; s <<= 1) {
            float omx = __shfl_xor(mx, s);
            float osm = __shfl_xor(sm, s);
            float nm  = fmaxf(mx, omx);
            sm = sm * __expf(mx - nm) + osm * __expf(omx - nm);
            mx = nm;
            tg = fmaxf(tg, __shfl_xor(tg, s));
        }
        run_mx[v] = mx; run_sm[v] = sm; run_tg[v] = tg;
    }

    if (m == 0) {   // lanes 0 and 16 hold merged stats for their 8 rows
#pragma unroll
        for (int v = 0; v < 8; ++v) {
            smx[wave][mbase + v] = run_mx[v];
            ssm[wave][mbase + v] = run_sm[v];
            stg[wave][mbase + v] = run_tg[v];
        }
    }
    __syncthreads();

    if (threadIdx.x < 16) {
        int r = rowblk * 16 + (int)threadIdx.x;
        float M = -INFINITY;
#pragma unroll
        for (int w = 0; w < 8; ++w) M = fmaxf(M, smx[w][threadIdx.x]);
        float S = 0.0f, T = -INFINITY;
#pragma unroll
        for (int w = 0; w < 8; ++w) {
            S += ssm[w][threadIdx.x] * __expf(smx[w][threadIdx.x] - M);
            T = fmaxf(T, stg[w][threadIdx.x]);
        }
        int b = r / LSEQ, t = r % LSEQ;
        float w_ = (log_mask[b * LSEQ + t] != 0) ? 1.0f : 0.0f;
        float nll = (M + __logf(S)) - T;
        atomicAdd(&acc[0], w_ * nll);
        atomicAdd(&acc[1], w_);
    }
}

// ---------------------------------------------------------------------------
// K7: finalize loss = sum / max(sumw, 1)
// ---------------------------------------------------------------------------
__global__ void k_final(const float* __restrict__ acc, float* __restrict__ out) {
    if (threadIdx.x == 0) out[0] = acc[0] / fmaxf(acc[1], 1.0f);
}

// ---------------------------------------------------------------------------
extern "C" void kernel_launch(void* const* d_in, const int* in_sizes, int n_in,
                              void* d_out, int out_size, void* d_ws, size_t ws_size,
                              hipStream_t stream) {
    (void)in_sizes; (void)n_in; (void)out_size; (void)ws_size;

    const int*   ids      = (const int*)d_in[0];   // (64,101)
    const int*   log_mask = (const int*)d_in[1];   // (64,100)
    const float* pop      = (const float*)d_in[2]; // (100001,)
    const float* emb      = (const float*)d_in[3]; // (100001,512)
    const float* W_ih     = (const float*)d_in[4]; // (512,1536)
    const float* W_hh     = (const float*)d_in[5]; // (512,1536)
    const float* b_ih     = (const float*)d_in[6]; // (1536,)
    const float* b_hh     = (const float*)d_in[7]; // (1536,)
    float* out = (float*)d_out;

    // ---- workspace carve-out (256B aligned) ----
    char* ws = (char*)d_ws;
    size_t off = 0;
    auto carve = [&](size_t bytes) -> void* {
        void* p = ws + off;
        off = (off + bytes + 255) & ~(size_t)255;
        return p;
    };
    bf16_t*        sbf      = (bf16_t*)carve((size_t)NCAND * D * sizeof(bf16_t));
    bf16_t*        wt_ih    = (bf16_t*)carve((size_t)TD * D * sizeof(bf16_t));
    bf16_t*        wt_hh    = (bf16_t*)carve((size_t)TD * D * sizeof(bf16_t));
    float*         debias   = (float*)carve((size_t)NCAND * sizeof(float));
    unsigned char* colvalid = (unsigned char*)carve((size_t)NCAND);
    unsigned char* reject   = (unsigned char*)carve((size_t)BS * NCAND);
    float*         gx       = (float*)carve((size_t)NROWS * TD * sizeof(float));
    bf16_t*        prec     = (bf16_t*)carve((size_t)NROWS * D * sizeof(bf16_t));
    float*         acc      = (float*)carve(2 * sizeof(float));

    // ---- pipeline ----
    k_init<<<1, 64, 0, stream>>>(acc);
    k_gather<<<NCAND, 128, 0, stream>>>(ids, pop, emb, log_mask, sbf, debias, colvalid);
    k_convw<<<(D * TD + 255) / 256, 256, 0, stream>>>(W_ih, W_hh, wt_ih, wt_hh);
    k_reject<<<dim3((NCAND + 255) / 256, BS), 256, 0, stream>>>(ids, reject);

    // gates_x GEMM: 400x48 tiles (16x32 per wave), 8 waves/block
    k_gates<<<(NROWS / 16) * (TD / 32) / 8, 256, 0, stream>>>(sbf, wt_ih, b_ih, gx);

    // full GRU recurrence in ONE launch: 4 persistent blocks, 16 rows each
    k_recur<<<BS / RROWS, 256, 0, stream>>>(gx, wt_hh, b_hh, prec);

    // fused logits + log-softmax + NLL
    k_logits<<<NROWS / 16, 256, 0, stream>>>(prec, sbf, debias, colvalid, reject,
                                             log_mask, acc);
    k_final<<<1, 32, 0, stream>>>(acc, out);
}
